// SelfAttention_34969623724257
// MI455X (gfx1250) — compile-verified, matrix operand order checked by reading, not run on hardware
//
#include <hip/hip_runtime.h>
#include <stdint.h>

#define DEV __device__ __forceinline__

// ---------------- types ----------------
typedef __attribute__((ext_vector_type(16))) __bf16         v16bf;
typedef __attribute__((ext_vector_type(16))) unsigned short v16u;
typedef __attribute__((ext_vector_type(8)))  unsigned short v8u;
typedef __attribute__((ext_vector_type(8)))  float          v8f;

static constexpr int Bb = 2, Ss = 2048, Dd = 1024, Hh = 16, HD = 64;

// ---------------- helpers ----------------
DEV unsigned short f32_to_bf16(float f) {
  unsigned int u = __builtin_bit_cast(unsigned int, f);
  u += 0x7fffu + ((u >> 16) & 1u);           // round-to-nearest-even
  return (unsigned short)(u >> 16);
}

DEV v8f wmma_bf16(v16bf a, v16bf b, v8f c) {
  // D = A(16x32 bf16) x B(32x16 bf16) + C(16x16 f32)
  return __builtin_amdgcn_wmma_f32_16x16x32_bf16(
      /*neg_a=*/false, a, /*neg_b=*/false, b,
      /*c_mod=*/(short)0, c, /*reuse_a=*/false, /*reuse_b=*/false);
}

// Load one 16x32 bf16 operand fragment. `p0` points at element [row0][k0] of a
// row-major [16 x stride] bf16 tile. Per ISA 7.12.2: lane holds row (lane&15);
// lanes 0-15 hold k = {0..7,16..23}, lanes 16-31 hold k = {8..15,24..31}
// -> two contiguous 16-byte runs per lane.
DEV v16bf load_frag(const unsigned short* p0, int stride, int lane) {
  const unsigned short* p = p0 + (lane & 15) * stride + ((lane >> 4) << 3);
  v8u lo = *(const v8u*)(p);
  v8u hi = *(const v8u*)(p + 16);
  v16u r = __builtin_shufflevector(lo, hi, 0,1,2,3,4,5,6,7,8,9,10,11,12,13,14,15);
  return __builtin_bit_cast(v16bf, r);
}

// CDNA5 async copy: per-lane 16B global -> LDS, tracked by ASYNCcnt (ISA 15.18.3).
DEV void async_b128_to_lds(unsigned lds_byte_off, const unsigned short* g) {
  unsigned long long ga = (unsigned long long)(uintptr_t)g;
  asm volatile("global_load_async_to_lds_b128 %0, %1, off"
               :: "v"(lds_byte_off), "v"(ga) : "memory");
}
DEV void wait_asynccnt0() { asm volatile("s_wait_asynccnt 0x0" ::: "memory"); }
DEV unsigned lds_off32(const void* p) { return (unsigned)(uintptr_t)p; }  // addr[31:0] = LDS offset

// Reductions within each 16-lane half of the wave32 (C-frag rows live per-half).
DEV float half16_max(float v) {
#pragma unroll
  for (int m = 1; m <= 8; m <<= 1) v = fmaxf(v, __shfl_xor(v, m, 32));
  return v;
}
DEV float half16_sum(float v) {
#pragma unroll
  for (int m = 1; m <= 8; m <<= 1) v += __shfl_xor(v, m, 32);
  return v;
}

// ---------------- weight conversion ----------------
__global__ __launch_bounds__(256) void cvt_f32_bf16_kernel(
    const float* __restrict__ src, unsigned short* __restrict__ dst, int n) {
  int i = blockIdx.x * 256 + threadIdx.x;
  if (i < n) dst[i] = f32_to_bf16(src[i]);
}

// ---------------- tiled WMMA GEMM ----------------
// C[M,N] = (A[M,K] @ W[K,N] + bias[N]) * alpha
// Block tile 256x128, BK=32, 256 threads = 8 wave32 in a 4x2 grid;
// wave tile 64x64 = 4x4 WMMA accumulators (16 WMMAs : 8 frag loads per k-step).
template <bool A_F32, bool OUT_F32>
__global__ __launch_bounds__(256) void gemm_wmma_kernel(
    const void* __restrict__ Aop, const unsigned short* __restrict__ W,
    const float* __restrict__ bias, void* __restrict__ Cout,
    int M, int N, int K, float alpha) {
  __shared__ alignas(16) unsigned short As[256 * 32];  // [m][k] bf16
  __shared__ alignas(16) unsigned short Bs[128 * 32];  // [n][k] bf16 (W tile transposed)

  const int tid  = threadIdx.x;
  const int lane = tid & 31;
  const int wid  = tid >> 5;
  const int row0 = blockIdx.y * 256;
  const int col0 = blockIdx.x * 128;
  const int wrow = (wid >> 1) * 64;   // 0,64,128,192
  const int wcol = (wid & 1) * 64;    // 0,64

  const float*          Af = (const float*)Aop;
  const unsigned short* Ab = (const unsigned short*)Aop;

  v8f c[4][4];
#pragma unroll
  for (int i = 0; i < 4; ++i)
#pragma unroll
    for (int j = 0; j < 4; ++j) c[i][j] = (v8f){};

  const unsigned as_base = lds_off32(&As[0]);

  for (int k0 = 0; k0 < K; k0 += 32) {
    // ---- stage A tile [256 x 32] ----
    if constexpr (A_F32) {
      // f32 -> bf16 conversion must pass through VGPRs
#pragma unroll 4
      for (int i = tid; i < 256 * 32; i += 256) {
        int r = i >> 5, k = i & 31;
        As[i] = f32_to_bf16(Af[(size_t)(row0 + r) * K + (k0 + k)]);
      }
    } else {
      // bf16 already: CDNA5 async global->LDS copy, 16B per lane, 1024 moves
#pragma unroll
      for (int t2 = 0; t2 < 4; ++t2) {
        int i = t2 * 256 + tid;            // 0..1023
        int r = i >> 2, seg = i & 3;       // 4 x 16B segments per 64B row
        async_b128_to_lds(as_base + i * 16,
                          Ab + (size_t)(row0 + r) * K + k0 + seg * 8);
      }
    }
    // ---- stage W tile transposed: Bs[n][k] = W[k0+k][col0+n] ----
#pragma unroll 4
    for (int i = tid; i < 128 * 32; i += 256) {
      int k = i >> 7, n = i & 127;
      Bs[n * 32 + k] = W[(size_t)(k0 + k) * N + (col0 + n)];
    }
    if constexpr (!A_F32) wait_asynccnt0();
    __syncthreads();

    v16bf a[4], b[4];
#pragma unroll
    for (int i = 0; i < 4; ++i) a[i] = load_frag(As + (wrow + 16 * i) * 32, 32, lane);
#pragma unroll
    for (int j = 0; j < 4; ++j) b[j] = load_frag(Bs + (wcol + 16 * j) * 32, 32, lane);
#pragma unroll
    for (int i = 0; i < 4; ++i)
#pragma unroll
      for (int j = 0; j < 4; ++j) c[i][j] = wmma_bf16(a[i], b[j], c[i][j]);
    __syncthreads();
  }

  // epilogue: C-frag layout row = r + 8*(lane>=16), col = lane&15
  const int hi = lane >> 4, ln = lane & 15;
#pragma unroll
  for (int i = 0; i < 4; ++i)
#pragma unroll
    for (int j = 0; j < 4; ++j) {
      const int gcol = col0 + wcol + 16 * j + ln;
      const float bv = bias[gcol];
#pragma unroll
      for (int r = 0; r < 8; ++r) {
        const int grow = row0 + wrow + 16 * i + r + 8 * hi;
        const float v = (c[i][j][r] + bv) * alpha;
        if (OUT_F32)
          ((float*)Cout)[(size_t)grow * N + gcol] = v;
        else
          ((unsigned short*)Cout)[(size_t)grow * N + gcol] = f32_to_bf16(v);
      }
    }
}

// ---------------- flash attention ----------------
// grid = (S/128, B*H); block = 256 (8 waves). Each wave owns 16 q rows.
// Q/K/V/O are bf16 in [B,S,D] layout; head h occupies cols [h*64, h*64+64).
// Q was pre-scaled by 1/sqrt(HD) in its projection epilogue.
__global__ __launch_bounds__(256) void flash_attn_kernel(
    const unsigned short* __restrict__ Qb, const unsigned short* __restrict__ Kb,
    const unsigned short* __restrict__ Vb, unsigned short* __restrict__ Ob) {
  __shared__ alignas(16) unsigned short Ks[64 * 64];      // [key][d]  (async-filled)
  __shared__ alignas(16) unsigned short VsT[64 * 64];     // [d][key]  (VALU transpose)
  __shared__ alignas(16) unsigned short Ps[8][16 * 64];   // per-wave P tile [row][key]

  const int tid  = threadIdx.x;
  const int lane = tid & 31;
  const int wid  = tid >> 5;
  const int hi   = lane >> 4, ln = lane & 15;
  const int b    = blockIdx.y >> 4;
  const int h    = blockIdx.y & 15;
  const int qrow = blockIdx.x * 128 + wid * 16;
  const size_t hoff = (size_t)h * HD;

  // Q fragments (2 k-steps of 32 across HD=64), loaded once from global bf16
  v16bf aq[2];
#pragma unroll
  for (int k0 = 0; k0 < 2; ++k0)
    aq[k0] = load_frag(Qb + ((size_t)b * Ss + qrow) * Dd + hoff + 32 * k0, Dd, lane);

  v8f o[4] = {};                 // O accumulator, 16 x 64 per wave
  float mrow[8], lrow[8];        // per-lane: rows j + 8*hi
#pragma unroll
  for (int j = 0; j < 8; ++j) { mrow[j] = -1e30f; lrow[j] = 0.0f; }

  const unsigned ks_base = lds_off32(&Ks[0]);

  for (int kb = 0; kb < Ss; kb += 64) {
    // K tile [key][d]: natural layout -> async global->LDS (512 x 16B moves),
    // overlapped with the VALU transpose staging of V below.
#pragma unroll
    for (int t2 = 0; t2 < 2; ++t2) {
      int i = t2 * 256 + tid;            // 0..511
      int r = i >> 3, seg = i & 7;       // 8 x 16B segments per 128B row
      async_b128_to_lds(ks_base + i * 16,
                        Kb + ((size_t)b * Ss + kb + r) * Dd + hoff + seg * 8);
    }
    // V tile transposed [d][key] (needed for B-fragment reads of P@V)
#pragma unroll 4
    for (int i = tid; i < 64 * 64; i += 256) {
      int r = i >> 6, d = i & 63;
      VsT[d * 64 + r] = Vb[((size_t)b * Ss + kb + r) * Dd + hoff + d];
    }
    wait_asynccnt0();
    __syncthreads();

    // scores S = Q K^T  (4 n-tiles of 16 keys)
    v8f s[4];
#pragma unroll
    for (int t = 0; t < 4; ++t) {
      v8f acc = (v8f){};
      acc  = wmma_bf16(aq[0], load_frag(Ks + 16 * t * 64,      64, lane), acc);
      s[t] = wmma_bf16(aq[1], load_frag(Ks + 16 * t * 64 + 32, 64, lane), acc);
    }

    // online softmax: row max, rescale, exp, row sum
    float mnew[8], alph[8], rs[8];
#pragma unroll
    for (int j = 0; j < 8; ++j) {
      float mx = fmaxf(fmaxf(s[0][j], s[1][j]), fmaxf(s[2][j], s[3][j]));
      mx = half16_max(mx);
      mnew[j] = fmaxf(mrow[j], mx);
      alph[j] = __expf(mrow[j] - mnew[j]);
      mrow[j] = mnew[j];
      rs[j] = 0.0f;
    }
#pragma unroll
    for (int t = 0; t < 4; ++t)
#pragma unroll
      for (int j = 0; j < 8; ++j) {
        float p = __expf(s[t][j] - mnew[j]);
        rs[j] += p;
        Ps[wid][(j + 8 * hi) * 64 + 16 * t + ln] = f32_to_bf16(p);
      }
#pragma unroll
    for (int j = 0; j < 8; ++j) {
      lrow[j] = lrow[j] * alph[j] + half16_sum(rs[j]);
#pragma unroll
      for (int t = 0; t < 4; ++t) o[t][j] *= alph[j];
    }

    // P store -> A-frag reload is an intra-wave LDS RAW: CDNA5 split counter
    asm volatile("s_wait_dscnt 0x0" ::: "memory");

    // O += P V   (2 k-steps over the 64 keys, 4 output-dim tiles)
#pragma unroll
    for (int k0 = 0; k0 < 2; ++k0) {
      v16bf ap = load_frag(&Ps[wid][0] + 32 * k0, 64, lane);
#pragma unroll
      for (int t = 0; t < 4; ++t)
        o[t] = wmma_bf16(ap, load_frag(VsT + 16 * t * 64 + 32 * k0, 64, lane), o[t]);
    }
    __syncthreads();  // protect Ks/VsT for next iteration
  }

  // normalize and store bf16 [B,S,D]
#pragma unroll
  for (int j = 0; j < 8; ++j) {
    const float inv = 1.0f / lrow[j];
    const int grow = qrow + j + 8 * hi;
#pragma unroll
    for (int t = 0; t < 4; ++t)
      Ob[((size_t)b * Ss + grow) * Dd + hoff + 16 * t + ln] = f32_to_bf16(o[t][j] * inv);
  }
}

// ---------------- launcher ----------------
extern "C" void kernel_launch(void* const* d_in, const int* in_sizes, int n_in,
                              void* d_out, int out_size, void* d_ws, size_t ws_size,
                              hipStream_t stream) {
  (void)in_sizes; (void)n_in; (void)out_size; (void)ws_size;
  const float* query = (const float*)d_in[0];
  const float* key   = (const float*)d_in[1];
  const float* value = (const float*)d_in[2];
  const float* Wq = (const float*)d_in[3];  const float* bq = (const float*)d_in[4];
  const float* Wk = (const float*)d_in[5];  const float* bk = (const float*)d_in[6];
  const float* Wv = (const float*)d_in[7];  const float* bv = (const float*)d_in[8];
  const float* Wo = (const float*)d_in[9];  const float* bo = (const float*)d_in[10];

  // workspace layout (bf16 buffers)
  const size_t WMAT = (size_t)Dd * Dd;       // 1M elems
  const size_t ACT  = (size_t)Bb * Ss * Dd;  // 4M elems
  char* ws = (char*)d_ws;
  size_t off = 0;
  unsigned short* Wq_b = (unsigned short*)(ws + off); off += WMAT * 2;
  unsigned short* Wk_b = (unsigned short*)(ws + off); off += WMAT * 2;
  unsigned short* Wv_b = (unsigned short*)(ws + off); off += WMAT * 2;
  unsigned short* Wo_b = (unsigned short*)(ws + off); off += WMAT * 2;
  unsigned short* q_b  = (unsigned short*)(ws + off); off += ACT * 2;
  unsigned short* k_b  = (unsigned short*)(ws + off); off += ACT * 2;
  unsigned short* v_b  = (unsigned short*)(ws + off); off += ACT * 2;
  unsigned short* a_b  = (unsigned short*)(ws + off); off += ACT * 2;

  // 1) weights f32 -> bf16
  const int nW = (int)WMAT;
  cvt_f32_bf16_kernel<<<(nW + 255) / 256, 256, 0, stream>>>(Wq, Wq_b, nW);
  cvt_f32_bf16_kernel<<<(nW + 255) / 256, 256, 0, stream>>>(Wk, Wk_b, nW);
  cvt_f32_bf16_kernel<<<(nW + 255) / 256, 256, 0, stream>>>(Wv, Wv_b, nW);
  cvt_f32_bf16_kernel<<<(nW + 255) / 256, 256, 0, stream>>>(Wo, Wo_b, nW);

  // 2) projections (M=4096, N=K=1024); fold 1/sqrt(64) into Q
  const int M = Bb * Ss, N = Dd, K = Dd;
  dim3 ggrid(N / 128, M / 256);
  gemm_wmma_kernel<true,  false><<<ggrid, 256, 0, stream>>>(query, Wq_b, bq, q_b, M, N, K, 0.125f);
  gemm_wmma_kernel<true,  false><<<ggrid, 256, 0, stream>>>(key,   Wk_b, bk, k_b, M, N, K, 1.0f);
  gemm_wmma_kernel<true,  false><<<ggrid, 256, 0, stream>>>(value, Wv_b, bv, v_b, M, N, K, 1.0f);

  // 3) flash attention over 16 heads x 2 batches x 16 q-blocks
  flash_attn_kernel<<<dim3(Ss / 128, Bb * Hh), 256, 0, stream>>>(q_b, k_b, v_b, a_b);

  // 4) output projection (bf16 A staged via async-to-LDS) -> f32 d_out
  gemm_wmma_kernel<false, true><<<ggrid, 256, 0, stream>>>(a_b, Wo_b, bo, (float*)d_out, M, N, K, 1.0f);
}
